// InterAgg_87789131530588
// MI455X (gfx1250) — matrix operand order repaired and checked by reading, not compiled, not run on hardware
//
#include <hip/hip_runtime.h>
#include <hip/hip_bf16.h>

// Problem constants (from reference)
#define NN   100000   // N nodes
#define F    256      // feature dim
#define E    128      // embed dim
#define B    8192     // batch
#define D    32       // neighbors per relation
#define KSEL 16       // top-k
#define TILE_M 16
#define CATW (3*E + F)   // 640
#define APAD 16          // bf16 LDS row padding (32B) keeps 16B alignment

typedef __attribute__((ext_vector_type(16))) __bf16 v16bf;
typedef __attribute__((ext_vector_type(8)))  __bf16 v8bf;
typedef __attribute__((ext_vector_type(8)))  float  v8f;

// ---------------------------------------------------------------------------
// s1[i] = dot(features[i,:], wc[:,1]) + bc[1]   (one wave32 per row)
// ---------------------------------------------------------------------------
__global__ void s1_kernel(const float* __restrict__ features,
                          const float* __restrict__ wc,
                          const float* __restrict__ bc,
                          float* __restrict__ s1) {
    const int row  = blockIdx.x * 8 + (threadIdx.x >> 5);
    const int lane = threadIdx.x & 31;
    if (row >= NN) return;
    const float* f = features + (size_t)row * F;
    float a = 0.f;
    #pragma unroll
    for (int j = lane; j < F; j += 32) a += f[j] * wc[j * 2 + 1];
    #pragma unroll
    for (int off = 16; off > 0; off >>= 1) a += __shfl_xor(a, off, 32);
    if (lane == 0) s1[row] = a + bc[1];
}

// ---------------------------------------------------------------------------
// center_scores[b,:] = features[nodes[b],:] @ wc + bc   (one wave32 per b)
// ---------------------------------------------------------------------------
__global__ void center_kernel(const float* __restrict__ features,
                              const int* __restrict__ nodes,
                              const float* __restrict__ wc,
                              const float* __restrict__ bc,
                              float* __restrict__ out_center) {
    const int b    = blockIdx.x * 8 + (threadIdx.x >> 5);
    const int lane = threadIdx.x & 31;
    if (b >= B) return;
    const float* f = features + (size_t)nodes[b] * F;
    float a0 = 0.f, a1 = 0.f;
    #pragma unroll
    for (int j = lane; j < F; j += 32) {
        const float fv = f[j];
        a0 += fv * wc[j * 2 + 0];
        a1 += fv * wc[j * 2 + 1];
    }
    #pragma unroll
    for (int off = 16; off > 0; off >>= 1) {
        a0 += __shfl_xor(a0, off, 32);
        a1 += __shfl_xor(a1, off, 32);
    }
    if (lane == 0) {
        out_center[b * 2 + 0] = a0 + bc[0];
        out_center[b * 2 + 1] = a1 + bc[1];
    }
}

// ---------------------------------------------------------------------------
// Transpose + bf16-pack a (Kdim x E) f32 weight matrix into (E x Kdim) bf16.
// Output writes are coalesced (consecutive k per thread block).
// ---------------------------------------------------------------------------
__global__ void pack_weight_kernel(const float* __restrict__ w,  // Kdim x E
                                   __bf16* __restrict__ wt,      // E x Kdim
                                   int Kdim) {
    const int idx = blockIdx.x * 256 + threadIdx.x;
    if (idx >= Kdim * E) return;
    const int n = idx / Kdim;
    const int k = idx - n * Kdim;
    wt[(size_t)n * Kdim + k] = (__bf16)w[(size_t)k * E + n];
}

// ---------------------------------------------------------------------------
// Build a v16bf A-fragment from a bf16 LDS row (contiguous layout:
// lanes 0-15 hold K in {kk..kk+15}, lanes 16-31 hold
// K in {kk+16..kk+31}).  Two ds_load_b128 per lane.
// ---------------------------------------------------------------------------
__device__ __forceinline__ v16bf load_a_frag(const __bf16* row, int kk, int kA) {
    const v8bf lo = *(const v8bf*)(row + kk + kA);
    const v8bf hi = *(const v8bf*)(row + kk + 8 + kA);
    return __builtin_shufflevector(lo, hi, 0, 1, 2, 3, 4, 5, 6, 7,
                                   8, 9, 10, 11, 12, 13, 14, 15);
}

// ---------------------------------------------------------------------------
// Fused per-relation kernel: top-K select -> gather-mean into bf16 LDS tile ->
// WMMA GEMM (16xF tile @ packed bf16 FxE weights) -> ReLU -> r_out (B x E).
// Block = 256 threads = 8 waves; block handles 16 batch rows; wave w owns
// output columns [w*16, w*16+16).
// ---------------------------------------------------------------------------
__global__ void relation_kernel(const float* __restrict__ features,
                                const int* __restrict__ nodes,
                                const int* __restrict__ neighs,   // B x D
                                const float* __restrict__ s1,     // N
                                const __bf16* __restrict__ wt,    // E x F (packed)
                                float* __restrict__ r_out) {      // B x E
    __shared__ __bf16 Asm[TILE_M][F + APAD];   // mean feature tile (bf16)
    __shared__ int    sel[TILE_M][KSEL];

    const int tid  = threadIdx.x;
    const int lane = tid & 31;
    const int wave = tid >> 5;
    const int b0   = blockIdx.x * TILE_M;

    // ---- Phase 1: top-K (smallest |s1[n]-c|) selection, 1 thread / row ----
    if (tid < TILE_M) {
        const int b = b0 + tid;
        const float c = s1[nodes[b]];
        float dist[D];
        #pragma unroll
        for (int d = 0; d < D; ++d)
            dist[d] = fabsf(s1[neighs[(size_t)b * D + d]] - c);
        int cnt = 0;
        for (int d = 0; d < D && cnt < KSEL; ++d) {
            int rank = 0;
            #pragma unroll
            for (int j = 0; j < D; ++j)
                rank += (dist[j] < dist[d]) || (dist[j] == dist[d] && j < d);
            if (rank < KSEL) sel[tid][cnt++] = neighs[(size_t)b * D + d];
        }
    }
    __syncthreads();

    // ---- Phase 2: mean of K gathered rows -> bf16 LDS tile (coalesced) ----
    const int col = tid;   // blockDim.x == F == 256
    float acc[TILE_M];
    #pragma unroll
    for (int m = 0; m < TILE_M; ++m) acc[m] = 0.f;
    for (int k = 0; k < KSEL; ++k) {
        #pragma unroll
        for (int m = 0; m < TILE_M; ++m) {
            const int row = sel[m][k];
            if (k + 1 < KSEL)   // prefetch next gathered row (L2 -> WGP$)
                __builtin_prefetch(&features[(size_t)sel[m][k + 1] * F + col], 0, 0);
            acc[m] += features[(size_t)row * F + col];
        }
    }
    #pragma unroll
    for (int m = 0; m < TILE_M; ++m)
        Asm[m][col] = (__bf16)(acc[m] * (1.0f / KSEL));
    __syncthreads();

    // ---- Phase 3: WMMA GEMM over K=F, bf16 inputs, f32 accumulate ----
    v8f cfrag = {};
    const int mrow = lane & 15;
    const int kA   = (lane < 16) ? 0 : 16;   // A k-base (contiguous layout)
    const int kB   = (lane < 16) ? 0 : 16;   // B k-base
    const int ncol = wave * 16 + (lane & 15);
    const __bf16* arow = &Asm[mrow][0];
    const __bf16* bcol = wt + (size_t)ncol * F;
    #pragma unroll
    for (int kk = 0; kk < F; kk += 32) {
        const v16bf a  = load_a_frag(arow, kk, kA);
        const v16bf bf = *(const v16bf*)(bcol + kk + kB);  // 2x global_load_b128
        cfrag = __builtin_amdgcn_wmma_f32_16x16x32_bf16(
            false, a, false, bf, (short)0, cfrag, false, false);
    }

    // ---- ReLU + store (C layout: lane<16 -> M=v, lane>=16 -> M=8+v) ----
    const int mbase = (lane < 16) ? 0 : 8;
    #pragma unroll
    for (int v = 0; v < 8; ++v) {
        float val = cfrag[v];
        val = val > 0.f ? val : 0.f;
        r_out[(size_t)(b0 + mbase + v) * E + ncol] = val;
    }
}

// ---------------------------------------------------------------------------
// Final kernel: cat = [features[nodes], r1, r2, r3] staged as bf16 (16 x 640),
// WMMA GEMM @ packed weight (E x 640 bf16), ReLU, transposed store (E x B).
// ---------------------------------------------------------------------------
__global__ void final_kernel(const float* __restrict__ features,
                             const int* __restrict__ nodes,
                             const float* __restrict__ r1,
                             const float* __restrict__ r2,
                             const float* __restrict__ r3,
                             const __bf16* __restrict__ wt,      // E x CATW (packed)
                             float* __restrict__ out) {          // E x B
    __shared__ __bf16 Csm[TILE_M][CATW + APAD];

    const int tid  = threadIdx.x;
    const int lane = tid & 31;
    const int wave = tid >> 5;
    const int b0   = blockIdx.x * TILE_M;

    // ---- Stage cat tile into LDS as bf16 ----
    for (int idx = tid; idx < TILE_M * CATW; idx += 256) {
        const int m   = idx / CATW;
        const int col = idx - m * CATW;
        const int b   = b0 + m;
        float v;
        if (col < F)            v = features[(size_t)nodes[b] * F + col];
        else if (col < F + E)   v = r1[(size_t)b * E + (col - F)];
        else if (col < F + 2*E) v = r2[(size_t)b * E + (col - F - E)];
        else                    v = r3[(size_t)b * E + (col - F - 2*E)];
        Csm[m][col] = (__bf16)v;
    }
    __syncthreads();

    // ---- WMMA GEMM over K = 640 ----
    v8f cfrag = {};
    const int mrow = lane & 15;
    const int kA   = (lane < 16) ? 0 : 16;
    const int kB   = (lane < 16) ? 0 : 16;
    const int ncol = wave * 16 + (lane & 15);
    const __bf16* arow = &Csm[mrow][0];
    const __bf16* bcol = wt + (size_t)ncol * CATW;
    #pragma unroll 4
    for (int kk = 0; kk < CATW; kk += 32) {
        const v16bf a  = load_a_frag(arow, kk, kA);
        const v16bf bf = *(const v16bf*)(bcol + kk + kB);
        cfrag = __builtin_amdgcn_wmma_f32_16x16x32_bf16(
            false, a, false, bf, (short)0, cfrag, false, false);
    }

    // ---- ReLU + transposed store: out[e*B + b] ----
    const int mbase = (lane < 16) ? 0 : 8;
    #pragma unroll
    for (int v = 0; v < 8; ++v) {
        float val = cfrag[v];
        val = val > 0.f ? val : 0.f;
        out[(size_t)ncol * B + (b0 + mbase + v)] = val;
    }
}

// ---------------------------------------------------------------------------
extern "C" void kernel_launch(void* const* d_in, const int* in_sizes, int n_in,
                              void* d_out, int out_size, void* d_ws, size_t ws_size,
                              hipStream_t stream) {
    (void)in_sizes; (void)n_in; (void)out_size; (void)ws_size;

    const float* features = (const float*)d_in[0];
    const int*   nodes    = (const int*)d_in[1];
    const int*   n1       = (const int*)d_in[2];
    const int*   n2       = (const int*)d_in[3];
    const int*   n3       = (const int*)d_in[4];
    const float* wc       = (const float*)d_in[5];
    const float* bc       = (const float*)d_in[6];
    const float* w1       = (const float*)d_in[7];
    const float* w2       = (const float*)d_in[8];
    const float* w3       = (const float*)d_in[9];
    const float* weight   = (const float*)d_in[10];

    float* out          = (float*)d_out;
    float* out_combined = out;                   // E x B
    float* out_center   = out + (size_t)E * B;   // B x 2

    // Workspace layout (all offsets 256B-aligned):
    //   [0, 512K)        s1            (NN f32)
    //   [512K, +4M)x3    r1, r2, r3    (B x E f32 each)
    //   then bf16 packed weights: w1t/w2t/w3t (E x F, 64KB each), wt (E x CATW)
    char*   ws  = (char*)d_ws;
    float*  s1  = (float*)ws;
    float*  r1  = (float*)(ws + (1u << 19));
    float*  r2  = r1 + (size_t)B * E;
    float*  r3  = r2 + (size_t)B * E;
    char*   wp  = ws + (1u << 19) + 3u * ((size_t)B * E * 4);
    __bf16* w1t = (__bf16*)(wp);
    __bf16* w2t = (__bf16*)(wp + 1 * (size_t)E * F * 2);
    __bf16* w3t = (__bf16*)(wp + 2 * (size_t)E * F * 2);
    __bf16* wtc = (__bf16*)(wp + 3 * (size_t)E * F * 2);

    // Pack weights to transposed bf16 (one-time per launch, coalesced writes)
    pack_weight_kernel<<<(E * F    + 255) / 256, 256, 0, stream>>>(w1, w1t, F);
    pack_weight_kernel<<<(E * F    + 255) / 256, 256, 0, stream>>>(w2, w2t, F);
    pack_weight_kernel<<<(E * F    + 255) / 256, 256, 0, stream>>>(w3, w3t, F);
    pack_weight_kernel<<<(E * CATW + 255) / 256, 256, 0, stream>>>(weight, wtc, CATW);

    s1_kernel    <<<(NN + 7) / 8, 256, 0, stream>>>(features, wc, bc, s1);
    center_kernel<<<B / 8,        256, 0, stream>>>(features, nodes, wc, bc, out_center);
    relation_kernel<<<B / TILE_M, 256, 0, stream>>>(features, nodes, n1, s1, w1t, r1);
    relation_kernel<<<B / TILE_M, 256, 0, stream>>>(features, nodes, n2, s1, w2t, r2);
    relation_kernel<<<B / TILE_M, 256, 0, stream>>>(features, nodes, n3, s1, w3t, r3);
    final_kernel   <<<B / TILE_M, 256, 0, stream>>>(features, nodes, r1, r2, r3,
                                                    wtc, out_combined);
}
